// Model_62904091017828
// MI455X (gfx1250) — matrix operand order repaired and checked by reading, not compiled, not run on hardware
//
#include <hip/hip_runtime.h>
#include <cstddef>
#include <cstdint>

// ---------------------------------------------------------------------------
// TreeLSTM forest (B=1024 trees, 63 nodes, EMB=TREE=128, IN_FEAT=512, OUT=3)
// CDNA5 / gfx1250: wave32, v_wmma_f32_16x16x32_bf16 for all GEMMs.
// Round 2: register double-buffered B fragments in feats_gemm so the
// s_wait_loadcnt for K-step n+1 hides behind the 8 WMMAs of K-step n.
// ---------------------------------------------------------------------------

#define BT    1024
#define NNODE 63
#define NTOT  (BT * NNODE)   // 64512
#define KF    512            // IN_FEAT
#define TD    128            // TREE == EMB

typedef __attribute__((ext_vector_type(16))) __bf16 v16bf;
typedef __attribute__((ext_vector_type(8)))  __bf16 v8bf;
typedef __attribute__((ext_vector_type(8)))  float  v8f;

static __device__ __forceinline__ float bf2f(__bf16 b) {
    unsigned short s = __builtin_bit_cast(unsigned short, b);
    unsigned u = ((unsigned)s) << 16;
    return __builtin_bit_cast(float, u);
}
static __device__ __forceinline__ __bf16 f2bf(float f) {
    unsigned u = __builtin_bit_cast(unsigned, f);
    u += 0x7fffu + ((u >> 16) & 1u);          // round-to-nearest-even
    unsigned short s = (unsigned short)(u >> 16);
    return __builtin_bit_cast(__bf16, s);
}
static __device__ __forceinline__ v16bf ldfrag(const __bf16* lo, const __bf16* hi) {
    v8bf a = *(const v8bf*)lo;
    v8bf b = *(const v8bf*)hi;
    return __builtin_shufflevector(a, b, 0,1,2,3,4,5,6,7,8,9,10,11,12,13,14,15);
}
static __device__ __forceinline__ v8f wmma_bf(v16bf a, v16bf b, v8f c) {
    return __builtin_amdgcn_wmma_f32_16x16x32_bf16(false, a, false, b,
                                                   (short)0, c, false, false);
}
static __device__ __forceinline__ float sigm(float x) {
    return 1.0f / (1.0f + __expf(-x));
}

// ---------------------------------------------------------------------------
// Weight prep: build bf16 transposed weights once.
//   Wt [512 out][512 in] : out<384 -> W_iou col, out>=384 -> W_f col
//   Uti[384 out][128 in] : U_iou^T       Utf[128 out][128 in] : U_f^T
// ---------------------------------------------------------------------------
__global__ __launch_bounds__(256)
void prep_weights(const float* __restrict__ W_iou, const float* __restrict__ W_f,
                  const float* __restrict__ U_iou, const float* __restrict__ U_f,
                  __bf16* __restrict__ Wt, __bf16* __restrict__ Uti,
                  __bf16* __restrict__ Utf) {
    int idx = blockIdx.x * 256 + threadIdx.x;
    if (idx < 512 * 512) {
        int o = idx >> 9, k = idx & 511;
        float v = (o < 384) ? W_iou[(size_t)k * 384 + o]
                            : W_f[(size_t)k * 128 + (o - 384)];
        Wt[(size_t)o * 512 + k] = f2bf(v);
    } else if (idx < 512 * 512 + 384 * 128) {
        int r = idx - 512 * 512;
        int o = r >> 7, k = r & 127;
        Uti[(size_t)o * 128 + k] = f2bf(U_iou[(size_t)k * 384 + o]);
    } else if (idx < 512 * 512 + 384 * 128 + 128 * 128) {
        int r = idx - (512 * 512 + 384 * 128);
        int o = r >> 7, k = r & 127;
        Utf[(size_t)o * 128 + k] = f2bf(U_f[(size_t)k * 128 + o]);
    }
}

// ---------------------------------------------------------------------------
// Fused feature-assembly + GEMM:
//   X[n][0:384] = xiou = feats@W_iou + b_iou ;  X[n][384:512] = xf = feats@W_f + b_f
// feats row = [emb_table[tok], sort_table[srt], const, ctx] gathered straight
// into LDS as bf16.  Block tile 32 rows x 512 cols; 8 waves = 2 row strips x
// 4 col groups; each wave: 16x128 = 8 WMMA tiles, K loop 16 x 32 with
// ping-pong B-fragment prefetch (latency hidden behind 8 WMMAs per step).
// ---------------------------------------------------------------------------
__global__ __launch_bounds__(256)
void feats_gemm(const int* __restrict__ tok, const int* __restrict__ srt,
                const float* __restrict__ cconst, const float* __restrict__ ctx_root,
                const float* __restrict__ emb, const float* __restrict__ sortt,
                const __bf16* __restrict__ Wt, const float* __restrict__ b_iou,
                const float* __restrict__ b_f, __bf16* __restrict__ X) {
    __shared__ __bf16 As[32][KF];                       // 32 KB
    const int t    = threadIdx.x;
    const int row0 = blockIdx.x * 32;

    {   // cooperative gather: 2 threads per (row, 128-col segment)
        int r     = t >> 3;
        int seg   = (t >> 1) & 3;
        int piece = t & 1;
        int gn    = row0 + r;
        const float* src = nullptr;
        if (seg == 0)       src = emb   + (size_t)tok[gn] * TD;
        else if (seg == 1)  src = sortt + (size_t)srt[gn] * TD;
        else if (seg == 2)  src = cconst + (size_t)gn * TD;
        else if (ctx_root)  src = ctx_root + (size_t)(gn / NNODE) * TD;
        int base = seg * 128 + piece * 64;
        #pragma unroll
        for (int i = 0; i < 16; ++i) {
            float4 v = src ? ((const float4*)src)[piece * 16 + i]
                           : make_float4(0.f, 0.f, 0.f, 0.f);
            __bf16* d = &As[r][base + i * 4];
            d[0] = f2bf(v.x); d[1] = f2bf(v.y); d[2] = f2bf(v.z); d[3] = f2bf(v.w);
        }
    }
    __syncthreads();

    const int wave = t >> 5, lane = t & 31;
    const int wrow = wave & 1;          // 2 row strips of 16
    const int wcol = wave >> 1;         // 4 col groups of 128
    const int lrow = lane & 15, half = lane >> 4;

    // per-lane base pointer for this wave's B columns (tile ti at +ti*16 rows)
    const __bf16* bbase = Wt + (size_t)(wcol * 128 + lrow) * KF + half * 8;
    const __bf16* arow  = &As[wrow * 16 + lrow][half * 8];

    v8f   acc[8] = {};
    v16bf bf0[8], bf1[8];

    #pragma unroll
    for (int ti = 0; ti < 8; ++ti) {                    // prefetch K-step 0
        const __bf16* bp = bbase + (size_t)ti * 16 * KF;
        bf0[ti] = ldfrag(bp, bp + 16);
    }

    #pragma unroll 1
    for (int ks = 0; ks < 16; ks += 2) {
        const int kb0 = ks * 32, kb1 = kb0 + 32, kb2 = kb0 + 64;

        v16bf a0 = ldfrag(arow + kb0, arow + kb0 + 16);
        #pragma unroll
        for (int ti = 0; ti < 8; ++ti) {                // prefetch K-step ks+1
            const __bf16* bp = bbase + (size_t)ti * 16 * KF + kb1;
            bf1[ti] = ldfrag(bp, bp + 16);
        }
        #pragma unroll
        for (int ti = 0; ti < 8; ++ti)
            acc[ti] = wmma_bf(a0, bf0[ti], acc[ti]);

        v16bf a1 = ldfrag(arow + kb1, arow + kb1 + 16);
        #pragma unroll
        for (int ti = 0; ti < 8; ++ti) {                // prefetch K-step ks+2
            // last iteration overreads 64B past Wt into adjacent ws (unused)
            const __bf16* bp = bbase + (size_t)ti * 16 * KF + kb2;
            bf0[ti] = ldfrag(bp, bp + 16);
        }
        #pragma unroll
        for (int ti = 0; ti < 8; ++ti)
            acc[ti] = wmma_bf(a1, bf1[ti], acc[ti]);
    }

    const int orow = row0 + wrow * 16 + half * 8;
    #pragma unroll
    for (int ti = 0; ti < 8; ++ti) {
        int col    = wcol * 128 + ti * 16 + lrow;
        float bias = (col < 384) ? b_iou[col] : b_f[col - 384];
        #pragma unroll
        for (int r = 0; r < 8; ++r)
            X[(size_t)(orow + r) * KF + col] = f2bf(acc[ti][r] + bias);
    }
}

// ---------------------------------------------------------------------------
// Per-level recurrence GEMM over gathered h rows (bf16), K = 128.
//   mode 0: A row m = h[child0(m)] + h[child1(m)]  (for h_sum @ U_iou, Nout=384)
//   mode 1: A row m = h[child(m)]                  (for child_h @ U_f, Nout=128)
// Block tile 32 rows x 128 cols; wave: 16 rows x 32 cols = 2 tiles, 4 K steps
// (fully unrolled; compiler hoists the 8 B loads).
// ---------------------------------------------------------------------------
__global__ __launch_bounds__(256)
void level_gemm(const __bf16* __restrict__ h, const __bf16* __restrict__ Bt,
                float* __restrict__ out, int lvl, int mode, int Nout) {
    __shared__ __bf16 As[32][TD];                       // 8 KB
    const int t    = threadIdx.x;
    const int row0 = blockIdx.x * 32;
    const int cnt  = 32 >> lvl;                         // parents/tree at lvl
    const int offc = 64 - (64 >> (lvl - 1));            // child-level node offset

    {   // LDS fill (child-sum folded in for mode 0)
        int r = t >> 3, q = t & 7;
        int m = row0 + r;
        if (mode == 0) {
            int tree = m / cnt, j = m - tree * cnt;
            const __bf16* p0 = h + ((size_t)(tree * NNODE + offc + 2 * j)) * TD + q * 16;
            const __bf16* p1 = p0 + TD;
            v8bf a0 = *(const v8bf*)p0, a1 = *(const v8bf*)(p0 + 8);
            v8bf b0 = *(const v8bf*)p1, b1 = *(const v8bf*)(p1 + 8);
            __bf16* d = &As[r][q * 16];
            #pragma unroll
            for (int i = 0; i < 8; ++i) {
                d[i]     = f2bf(bf2f(a0[i]) + bf2f(b0[i]));
                d[8 + i] = f2bf(bf2f(a1[i]) + bf2f(b1[i]));
            }
        } else {
            int tc = 2 * cnt;
            int tree = m / tc, e = m - tree * tc;
            const __bf16* p = h + ((size_t)(tree * NNODE + offc + e)) * TD + q * 16;
            *(v8bf*)&As[r][q * 16]     = *(const v8bf*)p;
            *(v8bf*)&As[r][q * 16 + 8] = *(const v8bf*)(p + 8);
        }
    }
    __syncthreads();

    const int wave = t >> 5, lane = t & 31;
    const int wrow = wave & 1, wcol = wave >> 1;
    const int lrow = lane & 15, half = lane >> 4;
    const int colbase = blockIdx.y * 128 + wcol * 32;

    v8f acc0 = {}, acc1 = {};
    #pragma unroll
    for (int ks = 0; ks < 4; ++ks) {
        const int kb = ks * 32;
        v16bf a = ldfrag(&As[wrow * 16 + lrow][kb + half * 8],
                         &As[wrow * 16 + lrow][kb + 16 + half * 8]);
        const __bf16* bp0 = Bt + (size_t)(colbase + lrow) * TD + kb + half * 8;
        acc0 = wmma_bf(a, ldfrag(bp0, bp0 + 16), acc0);
        const __bf16* bp1 = Bt + (size_t)(colbase + 16 + lrow) * TD + kb + half * 8;
        acc1 = wmma_bf(a, ldfrag(bp1, bp1 + 16), acc1);
    }

    const int orow = row0 + wrow * 16 + half * 8;
    #pragma unroll
    for (int r = 0; r < 8; ++r) {
        out[(size_t)(orow + r) * Nout + colbase + lrow]      = acc0[r];
        out[(size_t)(orow + r) * Nout + colbase + 16 + lrow] = acc1[r];
    }
}

// ---------------------------------------------------------------------------
// Gate nonlinearities for one level.  One thread per (node, channel).
//   lvl 0 (leaves): c = sig(i)*tanh(u) ; h = sig(o)*tanh(c)
//   lvl>=1: add U-products and f-gated child cell sum.
// ---------------------------------------------------------------------------
__global__ __launch_bounds__(256)
void gate_kernel(const __bf16* __restrict__ X, const float* __restrict__ uiou,
                 const float* __restrict__ uf, float* __restrict__ c,
                 __bf16* __restrict__ h, float* __restrict__ root_out, int lvl) {
    const int cnt = 32 >> lvl;
    const int idx = blockIdx.x * 256 + threadIdx.x;
    const int ch  = idx & 127;
    const int m   = idx >> 7;                  // m = tree*cnt + j
    const int tree = m / cnt, j = m - tree * cnt;
    const int off  = 64 - (64 >> lvl);
    const int gn   = tree * NNODE + off + j;

    const __bf16* xp = X + (size_t)gn * KF;
    float xi = bf2f(xp[ch]);
    float xo = bf2f(xp[128 + ch]);
    float xu = bf2f(xp[256 + ch]);
    float csum = 0.0f;

    if (lvl > 0) {
        const float* up = uiou + (size_t)m * 384;
        xi += up[ch];
        xo += up[128 + ch];
        xu += up[256 + ch];
        float xfv = bf2f(xp[384 + ch]);
        float f0 = sigm(xfv + uf[(size_t)(2 * m) * TD + ch]);
        float f1 = sigm(xfv + uf[(size_t)(2 * m + 1) * TD + ch]);
        int offc = 64 - (64 >> (lvl - 1));
        int c0n  = tree * NNODE + offc + 2 * j;
        csum = f0 * c[(size_t)c0n * TD + ch] + f1 * c[(size_t)(c0n + 1) * TD + ch];
    }

    float cn = sigm(xi) * tanhf(xu) + csum;
    float hn = sigm(xo) * tanhf(cn);
    c[(size_t)gn * TD + ch] = cn;
    h[(size_t)gn * TD + ch] = f2bf(hn);
    if (root_out) root_out[(size_t)tree * TD + ch] = hn;   // lvl 5 only
}

// ---------------------------------------------------------------------------
// Final head: h = [ha, hb, dot(ha,hb)]; logits = relu(h@fc1+b1)@fc2+b2.
// One block (128 threads) per tree.
// ---------------------------------------------------------------------------
__global__ __launch_bounds__(128)
void fc_kernel(const float* __restrict__ ha, const float* __restrict__ hb,
               const float* __restrict__ fc1W, const float* __restrict__ fc1b,
               const float* __restrict__ fc2W, const float* __restrict__ fc2b,
               float* __restrict__ out) {
    __shared__ float sa[128], sb[128], sp[128], hid[128];
    const int b = blockIdx.x, t = threadIdx.x;
    sa[t] = ha[(size_t)b * 128 + t];
    sb[t] = hb[(size_t)b * 128 + t];
    sp[t] = sa[t] * sb[t];
    __syncthreads();
    for (int s = 64; s > 0; s >>= 1) {
        if (t < s) sp[t] += sp[t + s];
        __syncthreads();
    }
    const float dot = sp[0];
    float acc = fc1b[t];
    #pragma unroll 4
    for (int k = 0; k < 128; ++k) acc += sa[k] * fc1W[(size_t)k * 128 + t];
    #pragma unroll 4
    for (int k = 0; k < 128; ++k) acc += sb[k] * fc1W[(size_t)(128 + k) * 128 + t];
    acc += dot * fc1W[(size_t)256 * 128 + t];
    hid[t] = fmaxf(acc, 0.0f);
    __syncthreads();
    if (t < 3) {
        float o = fc2b[t];
        for (int jj = 0; jj < 128; ++jj) o += hid[jj] * fc2W[(size_t)jj * 3 + t];
        out[(size_t)b * 3 + t] = o;
    }
}

// ---------------------------------------------------------------------------
extern "C" void kernel_launch(void* const* d_in, const int* in_sizes, int n_in,
                              void* d_out, int out_size, void* d_ws, size_t ws_size,
                              hipStream_t stream) {
    (void)in_sizes; (void)n_in; (void)out_size; (void)ws_size;

    const int*   cube_tok = (const int*)d_in[0];
    const int*   cube_srt = (const int*)d_in[1];
    const float* cube_cst = (const float*)d_in[2];
    const int*   a_tok    = (const int*)d_in[3];
    const int*   a_srt    = (const int*)d_in[4];
    const float* a_cst    = (const float*)d_in[5];
    const int*   b_tok    = (const int*)d_in[6];
    const int*   b_srt    = (const int*)d_in[7];
    const float* b_cst    = (const float*)d_in[8];
    // d_in[9..11] = node_order / adj_parent / adj_child: structure is a perfect
    // binary tree laid out level-by-level, so indices are computed in closed form.
    const float* emb   = (const float*)d_in[12];
    const float* sortt = (const float*)d_in[13];
    const float* W_iou = (const float*)d_in[14];
    const float* b_iou = (const float*)d_in[15];
    const float* U_iou = (const float*)d_in[16];
    const float* W_f   = (const float*)d_in[17];
    const float* b_f   = (const float*)d_in[18];
    const float* U_f   = (const float*)d_in[19];
    const float* fc1W  = (const float*)d_in[20];
    const float* fc1b  = (const float*)d_in[21];
    const float* fc2W  = (const float*)d_in[22];
    const float* fc2b  = (const float*)d_in[23];

    // workspace carve-up (~160 MB, every buffer fully written before read)
    char* ws = (char*)d_ws;
    size_t off = 0;
    auto alloc = [&](size_t bytes) -> char* {
        char* p = ws + off;
        off = (off + bytes + 255) & ~(size_t)255;
        return p;
    };
    __bf16* Wt   = (__bf16*)alloc((size_t)512 * 512 * 2);
    __bf16* Uti  = (__bf16*)alloc((size_t)384 * 128 * 2);
    __bf16* Utf  = (__bf16*)alloc((size_t)128 * 128 * 2);
    __bf16* X    = (__bf16*)alloc((size_t)NTOT * KF * 2);
    __bf16* h    = (__bf16*)alloc((size_t)NTOT * TD * 2);
    float*  c    = (float*) alloc((size_t)NTOT * TD * 4);
    float*  hc   = (float*) alloc((size_t)BT * TD * 4);
    float*  ha   = (float*) alloc((size_t)BT * TD * 4);
    float*  hb   = (float*) alloc((size_t)BT * TD * 4);
    float*  uiou = (float*) alloc((size_t)16384 * 384 * 4);
    float*  uf   = (float*) alloc((size_t)32768 * 128 * 4);

    prep_weights<<<1280, 256, 0, stream>>>(W_iou, W_f, U_iou, U_f, Wt, Uti, Utf);

    auto run = [&](const int* tk, const int* sr, const float* cs,
                   const float* ctx, float* root) {
        feats_gemm<<<NTOT / 32, 256, 0, stream>>>(tk, sr, cs, ctx, emb, sortt,
                                                  Wt, b_iou, b_f, X);
        gate_kernel<<<(BT * 32 * 128) / 256, 256, 0, stream>>>(
            X, nullptr, nullptr, c, h, nullptr, 0);
        for (int lvl = 1; lvl <= 5; ++lvl) {
            int M = BT * (32 >> lvl);   // parents at this level
            level_gemm<<<dim3(M / 32, 3), 256, 0, stream>>>(h, Uti, uiou, lvl, 0, 384);
            level_gemm<<<dim3((2 * M) / 32, 1), 256, 0, stream>>>(h, Utf, uf, lvl, 1, 128);
            gate_kernel<<<(M * 128) / 256, 256, 0, stream>>>(
                X, uiou, uf, c, h, (lvl == 5) ? root : nullptr, lvl);
        }
    };

    run(cube_tok, cube_srt, cube_cst, nullptr, hc);  // context tree (ctx = 0)
    run(a_tok,    a_srt,    a_cst,    hc,      ha);  // tree A (ctx = h_c root)
    run(b_tok,    b_srt,    b_cst,    hc,      hb);  // tree B

    fc_kernel<<<BT, 128, 0, stream>>>(ha, hb, fc1W, fc1b, fc2W, fc2b, (float*)d_out);
}
